// LightweightOutputEmbedding_21414706938619
// MI455X (gfx1250) — matrix operand order repaired and checked by reading, not compiled, not run on hardware
//
#include <hip/hip_runtime.h>
#include <stdint.h>

#define VOCAB   32000
#define HIDDEN  512
#define KCAND   300
#define ROWS    4096                      // B*S = 2*2048
#define SCORES_ELEMS ((size_t)ROWS * VOCAB)   // 131,072,000 floats

typedef __attribute__((ext_vector_type(16))) __bf16 v16bf;
typedef __attribute__((ext_vector_type(8)))  float  v8f;

// float -> bf16 bits, round-to-nearest-even
__device__ __forceinline__ uint32_t f2bf(float f) {
    uint32_t u = __float_as_uint(f);
    return (u + 0x7fffu + ((u >> 16) & 1u)) >> 16;
}
__device__ __forceinline__ uint32_t pack2(float lo, float hi) {
    return f2bf(lo) | (f2bf(hi) << 16);
}

// ---------------------------------------------------------------------------
// Kernel 1: scores = H[4096x512] @ Wb[512x32000] in bf16 WMMA, fp32 accum.
// Block = 256 threads (8 wave32). Block tile 128(M) x 64(N).
// Wave tile 16(M) x 64(N): one A fragment reused by 4 B fragments -> 4 WMMA
// per K-chunk per wave. A staged fp32 via global_load_async_to_lds_b128
// (ASYNCcnt path); B converted to bf16 in regs and stored transposed to LDS.
// ---------------------------------------------------------------------------
__global__ __launch_bounds__(256)
void bin_scores_gemm(const float* __restrict__ H, const float* __restrict__ Wb,
                     float* __restrict__ scores) {
    __shared__ float          Afp[128 * 32];   // [m][k] fp32, row stride 32
    __shared__ unsigned short Blds[64 * 40];   // [n][k] bf16 (transposed), pad->40

    const int tid  = threadIdx.x;
    const int lane = tid & 31;
    const int wave = tid >> 5;                 // 0..7 -> 16-row strip
    const int rowBase = blockIdx.y * 128;
    const int colBase = blockIdx.x * 64;

    // staging assignments
    const int b_k  = tid >> 3;                 // 0..31 k-row of B chunk
    const int b_n0 = (tid & 7) << 3;           // 0,8,..,56

    // LDS byte offset of Afp (flat->LDS mapping uses addr[31:0], ISA 10.2)
    const uint32_t abase = (uint32_t)(uintptr_t)(&Afp[0]);

    v8f acc0 = {}, acc1 = {}, acc2 = {}, acc3 = {};

    for (int kk = 0; kk < HIDDEN; kk += 32) {
        // ---- stage A: 128 rows x 32 k fp32, async global->LDS (16B/lane x4)
#pragma unroll
        for (int i = 0; i < 4; ++i) {
            const int id  = i * 256 + tid;     // 0..1023 float4 units
            const int row = id >> 3;           // 0..127
            const int q   = id & 7;            // 0..7 float4 within row
            const uint64_t gaddr = (uint64_t)(uintptr_t)(
                H + (size_t)(rowBase + row) * HIDDEN + kk + (q << 2));
            const uint32_t laddr = abase + (uint32_t)(row * 128 + q * 16);
            asm volatile("global_load_async_to_lds_b128 %0, %1, off"
                         :: "v"(laddr), "v"(gaddr) : "memory");
        }

        // ---- stage B: 32 k x 64 n, fp32 -> bf16, transposed to [n][k] ----
        const float* wrow = Wb + (size_t)(kk + b_k) * VOCAB + colBase + b_n0;
        const float4 b0 = *reinterpret_cast<const float4*>(wrow);
        const float4 b1 = *reinterpret_cast<const float4*>(wrow + 4);
        Blds[(b_n0 + 0) * 40 + b_k] = (unsigned short)f2bf(b0.x);
        Blds[(b_n0 + 1) * 40 + b_k] = (unsigned short)f2bf(b0.y);
        Blds[(b_n0 + 2) * 40 + b_k] = (unsigned short)f2bf(b0.z);
        Blds[(b_n0 + 3) * 40 + b_k] = (unsigned short)f2bf(b0.w);
        Blds[(b_n0 + 4) * 40 + b_k] = (unsigned short)f2bf(b1.x);
        Blds[(b_n0 + 5) * 40 + b_k] = (unsigned short)f2bf(b1.y);
        Blds[(b_n0 + 6) * 40 + b_k] = (unsigned short)f2bf(b1.z);
        Blds[(b_n0 + 7) * 40 + b_k] = (unsigned short)f2bf(b1.w);

        if (kk + 32 < HIDDEN)                  // prefetch next W chunk row
            __builtin_prefetch(wrow + (size_t)VOCAB * 32, 0, 0);

        // drain this wave's async copies, then workgroup barrier
        asm volatile("s_wait_asynccnt 0x0" ::: "memory");
        __syncthreads();

        // ---- A fragment (16x32 bf16) from fp32 LDS: lane m = lane%16;
        //      half-wave 0 holds K 0-15, half-wave 1 holds K 16-31
        const int am  = wave * 16 + (lane & 15);
        const int akb = (lane < 16) ? 0 : 16;
        const float4 a0 = *reinterpret_cast<const float4*>(&Afp[am * 32 + akb]);
        const float4 a1 = *reinterpret_cast<const float4*>(&Afp[am * 32 + akb + 4]);
        const float4 a2 = *reinterpret_cast<const float4*>(&Afp[am * 32 + akb + 8]);
        const float4 a3 = *reinterpret_cast<const float4*>(&Afp[am * 32 + akb + 12]);
        union { uint32_t u[8]; v16bf v; } af;
        af.u[0] = pack2(a0.x, a0.y); af.u[1] = pack2(a0.z, a0.w);
        af.u[2] = pack2(a1.x, a1.y); af.u[3] = pack2(a1.z, a1.w);
        af.u[4] = pack2(a2.x, a2.y); af.u[5] = pack2(a2.z, a2.w);
        af.u[6] = pack2(a3.x, a3.y); af.u[7] = pack2(a3.z, a3.w);

        // ---- 4 B fragments (32x16 bf16): lane n = lane%16;
        //      half-wave 0 holds K 0-15, half-wave 1 holds K 16-31
        const int bn  = lane & 15;
        const int bkb = (lane < 16) ? 0 : 16;
        union { uint4 q[2]; v16bf v; } bf0, bf1, bf2, bf3;
        bf0.q[0] = *reinterpret_cast<const uint4*>(&Blds[(bn +  0) * 40 + bkb]);
        bf0.q[1] = *reinterpret_cast<const uint4*>(&Blds[(bn +  0) * 40 + bkb + 8]);
        bf1.q[0] = *reinterpret_cast<const uint4*>(&Blds[(bn + 16) * 40 + bkb]);
        bf1.q[1] = *reinterpret_cast<const uint4*>(&Blds[(bn + 16) * 40 + bkb + 8]);
        bf2.q[0] = *reinterpret_cast<const uint4*>(&Blds[(bn + 32) * 40 + bkb]);
        bf2.q[1] = *reinterpret_cast<const uint4*>(&Blds[(bn + 32) * 40 + bkb + 8]);
        bf3.q[0] = *reinterpret_cast<const uint4*>(&Blds[(bn + 48) * 40 + bkb]);
        bf3.q[1] = *reinterpret_cast<const uint4*>(&Blds[(bn + 48) * 40 + bkb + 8]);

        acc0 = __builtin_amdgcn_wmma_f32_16x16x32_bf16(false, af.v, false, bf0.v,
                                                       (short)0, acc0, false, false);
        acc1 = __builtin_amdgcn_wmma_f32_16x16x32_bf16(false, af.v, false, bf1.v,
                                                       (short)0, acc1, false, false);
        acc2 = __builtin_amdgcn_wmma_f32_16x16x32_bf16(false, af.v, false, bf2.v,
                                                       (short)0, acc2, false, false);
        acc3 = __builtin_amdgcn_wmma_f32_16x16x32_bf16(false, af.v, false, bf3.v,
                                                       (short)0, acc3, false, false);

        __syncthreads();
    }

    // ---- epilogue: C layout = VGPR r: lanes 0-15 -> M=r, lanes 16-31 -> M=r+8
    const int mtop = (lane & 16) ? 8 : 0;
    const int ncol = colBase + (lane & 15);
    const int row0 = rowBase + wave * 16 + mtop;
#pragma unroll
    for (int r = 0; r < 8; ++r) {
        const size_t base = (size_t)(row0 + r) * VOCAB;
        scores[base + ncol +  0] = acc0[r];
        scores[base + ncol + 16] = acc1[r];
        scores[base + ncol + 32] = acc2[r];
        scores[base + ncol + 48] = acc3[r];
    }
}

// ---------------------------------------------------------------------------
// Kernel 2: per-row top-300 (descending value, ascending index on ties).
// One 256-thread block per row; iterative constrained arg-max on 64-bit keys.
// ---------------------------------------------------------------------------
__global__ __launch_bounds__(256)
void topk_rows(const float* __restrict__ scores, int* __restrict__ topk,
               float* __restrict__ idx_out) {
    __shared__ unsigned long long red[256];
    __shared__ unsigned long long prevk_sh;
    const int row = blockIdx.x;
    const int tid = threadIdx.x;
    const float* srow = scores + (size_t)row * VOCAB;

    if (tid == 0) prevk_sh = ~0ull;
    __syncthreads();

    for (int k = 0; k < KCAND; ++k) {
        const unsigned long long prev = prevk_sh;
        unsigned long long best = 0ull;
        for (int j = tid; j < VOCAB; j += 256) {
            uint32_t u = __float_as_uint(srow[j]);
            uint32_t s = (u & 0x80000000u) ? ~u : (u | 0x80000000u); // sortable
            unsigned long long key =
                ((unsigned long long)s << 32) | (uint32_t)(VOCAB - 1 - j);
            if (key < prev && key > best) best = key;
        }
        red[tid] = best;
        __syncthreads();
        for (int s = 128; s > 0; s >>= 1) {
            if (tid < s) {
                unsigned long long o = red[tid + s];
                if (o > red[tid]) red[tid] = o;
            }
            __syncthreads();
        }
        if (tid == 0) {
            unsigned long long b = red[0];
            int idx = (VOCAB - 1) - (int)(b & 0xffffffffu);
            topk[(size_t)row * KCAND + k]    = idx;
            idx_out[(size_t)row * KCAND + k] = (float)idx;
            prevk_sh = b;
        }
        __syncthreads();
    }
}

// ---------------------------------------------------------------------------
// Kernel 3: zero the sparse-logits region (it held the bf16-GEMM scores).
// ---------------------------------------------------------------------------
__global__ void zero_scores(float4* __restrict__ out) {
    size_t i = (size_t)blockIdx.x * blockDim.x + threadIdx.x;
    out[i] = make_float4(0.f, 0.f, 0.f, 0.f);
}

// ---------------------------------------------------------------------------
// Kernel 4: exact fp32 logits at the 300 selected columns, scattered store.
// One wave32 per (row, candidate): 512-MAC dot product + shuffle reduction.
// ---------------------------------------------------------------------------
__global__ __launch_bounds__(256)
void candidate_logits(const float* __restrict__ H, const float* __restrict__ Wo,
                      const int* __restrict__ topk, float* __restrict__ out) {
    const int lane = threadIdx.x & 31;
    const int wave = threadIdx.x >> 5;
    const long long pair = (long long)blockIdx.x * 8 + wave;
    if (pair >= (long long)ROWS * KCAND) return;
    const int row = (int)(pair / KCAND);
    const int k   = (int)(pair % KCAND);
    const int idx = topk[(size_t)row * KCAND + k];

    const float* h = H  + (size_t)row * HIDDEN;
    const float* w = Wo + idx;                 // column idx, stride VOCAB
    float sum = 0.f;
#pragma unroll
    for (int j = 0; j < HIDDEN / 32; ++j) {
        const int jj = j * 32 + lane;
        sum += h[jj] * w[(size_t)jj * VOCAB];
    }
#pragma unroll
    for (int off = 16; off > 0; off >>= 1)
        sum += __shfl_down(sum, off, 32);
    if (lane == 0) out[(size_t)row * VOCAB + idx] = sum;
}

// ---------------------------------------------------------------------------
extern "C" void kernel_launch(void* const* d_in, const int* in_sizes, int n_in,
                              void* d_out, int out_size, void* d_ws, size_t ws_size,
                              hipStream_t stream) {
    (void)in_sizes; (void)n_in; (void)out_size; (void)ws_size;
    const float* H  = (const float*)d_in[0];   // hidden_states [4096,512]
    const float* Wo = (const float*)d_in[1];   // W_out  [512,32000]
    const float* Wb = (const float*)d_in[2];   // W_bin  [512,32000]

    float* out     = (float*)d_out;
    float* scores  = out;                      // reuse logits region as scratch
    float* idx_out = out + SCORES_ELEMS;       // second output: indices (as fp32)
    int*   topk    = (int*)d_ws;               // 4096*300*4 B = 4.9 MB

    dim3 gGemm(VOCAB / 64, ROWS / 128);        // 500 x 32 blocks
    bin_scores_gemm<<<gGemm, 256, 0, stream>>>(H, Wb, scores);

    topk_rows<<<ROWS, 256, 0, stream>>>(scores, topk, idx_out);

    zero_scores<<<(unsigned)(SCORES_ELEMS / 4 / 256), 256, 0, stream>>>(
        (float4*)out);

    candidate_logits<<<(ROWS * KCAND) / 8, 256, 0, stream>>>(H, Wo, topk, out);
}
